// scene_graph_with_monet_62740882260662
// MI455X (gfx1250) — matrix operand order repaired and checked by reading, not compiled, not run on hardware
//
#include <hip/hip_runtime.h>
#include <hip/hip_bf16.h>

// ---------------------------------------------------------------------------
// Problem constants (from reference)
// ---------------------------------------------------------------------------
#define BB      4
#define CC      256
#define HWW     384          // H*W = 16*24
#define SS      11
#define S2      121
#define DD1     256
#define NPIX    (BB * HWW)   // 1536
#define KTOT    (DD1 * HWW)  // 98304
#define KCHUNKS (KTOT / 32)  // 3072
#define KSPLIT  8
#define CHUNKS_PER_SPLIT (KCHUNKS / KSPLIT) // 384
#define OBJ_OUT_ELEMS (BB * SS * DD1)       // 11264

typedef __bf16 bf16_t;
typedef __attribute__((ext_vector_type(16))) __bf16 v16bf;
typedef __attribute__((ext_vector_type(8)))  float   v8f;
typedef __attribute__((ext_vector_type(4)))  unsigned u32x4;

union Frag {
  unsigned u[8];
  u32x4    q[2];
  v16bf    v;
};

__device__ __forceinline__ unsigned pack_bf2(float lo, float hi) {
  unsigned short a = __builtin_bit_cast(unsigned short, (bf16_t)lo);
  unsigned short b = __builtin_bit_cast(unsigned short, (bf16_t)hi);
  return (unsigned)a | ((unsigned)b << 16);
}

// ---------------------------------------------------------------------------
// Kernel 0a: transpose+convert input (B,C,HW) -> X0h[pix][c] bf16
// ---------------------------------------------------------------------------
__global__ void k_cvt_in(const float* __restrict__ in, bf16_t* __restrict__ X0h) {
  int tid = blockIdx.x * 256 + threadIdx.x;          // 0 .. NPIX*CC-1
  int pix = tid >> 8;
  int ch  = tid & 255;
  int b   = pix / HWW;
  int hw  = pix % HWW;
  X0h[tid] = (bf16_t)in[((size_t)(b * CC + ch)) * HWW + hw];
}

// ---------------------------------------------------------------------------
// Kernel 0b: convert Wfc fp32 -> bf16 B-fragment layout
//   Wb[kc][nt][lane][v] (u32 = packed bf16 pair), element:
//   n = nt*16 + (lane&15), k = kc*32 + (lane>>4)*16 + 2v
// ---------------------------------------------------------------------------
__global__ void k_cvt_wfc(const float* __restrict__ Wfc, unsigned* __restrict__ Wb) {
  size_t tid = (size_t)blockIdx.x * 256 + threadIdx.x;  // 0 .. KCHUNKS*16*32*8-1
  int v    = (int)(tid & 7);
  int lane = (int)((tid >> 3) & 31);
  int nt   = (int)((tid >> 8) & 15);
  int kc   = (int)(tid >> 12);
  int n = nt * 16 + (lane & 15);
  int k = kc * 32 + ((lane >> 4) << 4) + (v << 1);
  const float* row = Wfc + (size_t)n * KTOT + k;
  Wb[tid] = pack_bf2(row[0], row[1]);
}

// ---------------------------------------------------------------------------
// Kernel 0c: cRel[o] = brf[o] + sum_k (Wrf[o,k] + Wrf[o,256+k]) * bof[k]
// ---------------------------------------------------------------------------
__global__ void k_crel(const float* __restrict__ Wrf, const float* __restrict__ brf,
                       const float* __restrict__ bof, float* __restrict__ cRel) {
  int o = threadIdx.x;
  const float* row = Wrf + (size_t)o * 896;
  float s = brf[o];
  for (int k = 0; k < 256; ++k) s += (row[k] + row[256 + k]) * bof[k];
  cRel[o] = s;
}

// ---------------------------------------------------------------------------
// Generic prep GEMM: one wave (32 thr) per 16x16 output tile.
//   C[m,n] = sum_k A[m,k]*W[n,k]   (A bf16 pixel-major, W fp32 row-major)
//   Cf (optional, fp32) stored feature-major [(b*256+n)*384+hw], acc optional
//   Ch (optional, bf16) stored pixel-major [m*ldch+n]
// NOTE: NPIX per batch (384) is a multiple of 16, so a tile never straddles
// a batch boundary -> bb / hw base are tile-uniform (no per-element division).
// ---------------------------------------------------------------------------
__global__ void k_gemm(const bf16_t* __restrict__ A, int lda,
                       const float* __restrict__ W, int ldb,
                       const float* __restrict__ bias,
                       float* __restrict__ Cf, int acc,
                       bf16_t* __restrict__ Ch, int ldch,
                       int K) {
  const int L = threadIdx.x;
  const int half = L >> 4;
  const int l16  = L & 15;
  const int rowbase = blockIdx.x * 16;
  const int mg = rowbase + l16;
  const int ng = blockIdx.y * 16 + l16;
  const bf16_t* arow = A + (size_t)mg * lda;
  const float*  wrow = W + (size_t)ng * ldb;
  v8f c = {0.f, 0.f, 0.f, 0.f, 0.f, 0.f, 0.f, 0.f};
  const int kch = K >> 5;
  for (int kc = 0; kc < kch; ++kc) {
    Frag a, b;
#pragma unroll
    for (int v = 0; v < 8; ++v) {   // A lane layout: K = half*16 + 2v
      int k = (kc << 5) + (half << 4) + (v << 1);
      a.u[v] = *(const unsigned*)(arow + k);
    }
#pragma unroll
    for (int v = 0; v < 8; ++v) {   // B lane layout: K = half*16 + 2v
      int k = (kc << 5) + (half << 4) + (v << 1);
      b.u[v] = pack_bf2(wrow[k], wrow[k + 1]);
    }
    c = __builtin_amdgcn_wmma_f32_16x16x32_bf16(false, a.v, false, b.v,
                                                (short)0, c, false, false);
  }
  // tile-uniform output decomposition
  const int bb     = rowbase / HWW;
  const int hwbase = rowbase - bb * HWW;
  const int col    = blockIdx.y * 16 + l16;
  float* cfcol = Cf ? (Cf + ((size_t)(bb * 256 + col)) * HWW + hwbase) : nullptr;
  bf16_t* chrow = Ch ? (Ch + (size_t)rowbase * ldch + col) : nullptr;
  const float bv = bias ? bias[col] : 0.f;
#pragma unroll
  for (int i = 0; i < 8; ++i) {
    const int rl = i + (half << 3);            // 0..15 row within tile
    float val = c[i] + bv;
    if (cfcol) {
      if (acc) val += cfcol[rl];
      cfcol[rl] = val;
    }
    if (chrow) chrow[(size_t)rl * ldch] = (bf16_t)val;
  }
}

// ---------------------------------------------------------------------------
// Main fused GEMM: rows = objects (OBJ) or pairs (!OBJ) of one batch b,
// A built on the fly = relu(const + coef-combination of pixel features),
// B = prepacked bf16 Wfc fragments.  K split 8 ways (blockIdx.y), partial
// 16x256 fp32 tiles written to Pout.  Double-buffered A tile: one barrier
// per K chunk.
// ---------------------------------------------------------------------------
template <bool OBJ>
__global__ void k_main(const float* __restrict__ masks,
                       const float* __restrict__ F1, const float* __restrict__ F2,
                       const float* __restrict__ F3, const float* __restrict__ F4,
                       const float* __restrict__ F5,
                       const float* __restrict__ cvec,
                       const unsigned* __restrict__ Wb,
                       float* __restrict__ Pout) {
  __shared__ float  maskL[SS * HWW];       // 16896 B
  __shared__ bf16_t atile[2][16 * 32];     // 2 KB, double buffered
  const int tid  = threadIdx.x;
  const int mt   = blockIdx.x;
  const int ks   = blockIdx.y;
  const int b    = OBJ ? mt : (mt >> 3);
  const int tile = OBJ ? 0  : (mt & 7);

  for (int i = tid; i < SS * HWW; i += 256)
    maskL[i] = masks[(size_t)b * (SS * HWW) + i];
  __syncthreads();

  const int w = tid >> 5;   // wave id == builder row group
  const int j = tid & 31;
  int  sOff[2], oOff[2];
  bool valid[2];
#pragma unroll
  for (int t = 0; t < 2; ++t) {
    int rr = w + 8 * t;
    if (OBJ) {
      valid[t] = rr < SS;
      sOff[t]  = valid[t] ? rr * HWW : 0;
      oOff[t]  = 0;
    } else {
      int p = tile * 16 + rr;
      valid[t] = p < S2;
      sOff[t]  = valid[t] ? (p / SS) * HWW : 0;
      oOff[t]  = valid[t] ? (p % SS) * HWW : 0;
    }
  }

  const int L = tid & 31;
  const int m = L & 15, half = L >> 4;
  const int nt0 = w * 2, nt1 = w * 2 + 1;
  v8f c0 = {0.f, 0.f, 0.f, 0.f, 0.f, 0.f, 0.f, 0.f};
  v8f c1 = {0.f, 0.f, 0.f, 0.f, 0.f, 0.f, 0.f, 0.f};
  const u32x4* wb4 = (const u32x4*)Wb;

  int d = ks * 32;  // feature channel for this K chunk
  int hw0 = 0;
  const size_t fbb = (size_t)b * 256 * HWW;
  const int kc0 = ks * CHUNKS_PER_SPLIT;
  const int kcEnd = kc0 + CHUNKS_PER_SPLIT;
  // incremental B-fragment indices (uint4 units); +1024 per chunk
  size_t wi0 = ((size_t)(kc0 * 16 + nt0) * 32 + L) * 2;
  size_t wi1 = ((size_t)(kc0 * 16 + nt1) * 32 + L) * 2;
  int buf = 0;
  for (int kc = kc0; kc < kcEnd; ++kc) {
    float cd = cvec[d];
    size_t fo = fbb + (size_t)d * HWW + hw0 + j;
    float f1 = F1[fo], f2 = F2[fo];
    float f3 = 0.f, f4 = 0.f, f5 = 0.f;
    if (!OBJ) { f3 = F3[fo]; f4 = F4[fo]; f5 = F5[fo]; }
#pragma unroll
    for (int t = 0; t < 2; ++t) {
      int rr = w + 8 * t;
      float val = 0.f;
      if (valid[t]) {
        float ms = maskL[sOff[t] + hw0 + j];
        if (OBJ) {
          val = cd + ms * f1 + ms * ms * f2;
        } else {
          float mo = maskL[oOff[t] + hw0 + j];
          val = cd + ms * f1 + mo * f2 + ms * ms * f3 + mo * mo * f4
                   + ms * mo * f5;
        }
        val = fmaxf(val, 0.f);
      }
      atile[buf][rr * 32 + j] = (bf16_t)val;
    }
    // prefetch next chunk's B fragments into cache (global_prefetch_b8)
    if (kc + 1 < kcEnd) {
      __builtin_prefetch((const void*)(wb4 + wi0 + 1024), 0, 1);
      __builtin_prefetch((const void*)(wb4 + wi1 + 1024), 0, 1);
    }
    __syncthreads();   // writes of atile[buf] visible to all waves
    Frag a, b0, b1;
    const u32x4* a4 = (const u32x4*)atile[buf];
    a.q[0] = a4[m * 4 + 2 * half];
    a.q[1] = a4[m * 4 + 2 * half + 1];
    b0.q[0] = wb4[wi0]; b0.q[1] = wb4[wi0 + 1];
    b1.q[0] = wb4[wi1]; b1.q[1] = wb4[wi1 + 1];
    c0 = __builtin_amdgcn_wmma_f32_16x16x32_bf16(false, a.v, false, b0.v,
                                                 (short)0, c0, false, false);
    c1 = __builtin_amdgcn_wmma_f32_16x16x32_bf16(false, a.v, false, b1.v,
                                                 (short)0, c1, false, false);
    wi0 += 1024;
    wi1 += 1024;
    buf ^= 1;
    hw0 += 32;
    if (hw0 == HWW) { hw0 = 0; ++d; }
  }

  float* pp = Pout + ((size_t)mt * KSPLIT + ks) * 4096;
#pragma unroll
  for (int i = 0; i < 8; ++i) {
    int rowl = i + (half << 3);
    pp[rowl * 256 + nt0 * 16 + m] = c0[i];
    pp[rowl * 256 + nt1 * 16 + m] = c1[i];
  }
}

// ---------------------------------------------------------------------------
// Finalize: reduce K-split partials, add bfc, L2-normalize rows, write out.
// ---------------------------------------------------------------------------
template <bool OBJ>
__global__ void k_fin(const float* __restrict__ P, const float* __restrict__ bfc,
                      float* __restrict__ out) {
  __shared__ float outL[16 * 256];
  __shared__ float red[16 * 16];
  __shared__ float nrm[16];
  const int tid = threadIdx.x;
  const int mt  = blockIdx.x;
  const float* pb = P + (size_t)mt * KSPLIT * 4096;
  for (int q = 0; q < 16; ++q) {
    int e = tid + q * 256;
    float s = bfc[e & 255];
#pragma unroll
    for (int ks = 0; ks < KSPLIT; ++ks) s += pb[ks * 4096 + e];
    outL[e] = s;
  }
  __syncthreads();
  {
    int row = tid >> 4, part = tid & 15;
    float s = 0.f;
#pragma unroll
    for (int q = 0; q < 16; ++q) {
      float v = outL[row * 256 + part * 16 + q];
      s += v * v;
    }
    red[tid] = s;
  }
  __syncthreads();
  if (tid < 16) {
    float s = 0.f;
#pragma unroll
    for (int q = 0; q < 16; ++q) s += red[tid * 16 + q];
    nrm[tid] = rsqrtf(s);
  }
  __syncthreads();
  for (int q = 0; q < 16; ++q) {
    int e = tid + q * 256;
    int row = e >> 8, col = e & 255;
    float v = outL[e] * nrm[row];
    if (OBJ) {
      if (row < SS) out[((size_t)(mt * SS + row)) * 256 + col] = v;
    } else {
      int bb = mt >> 3, tl = mt & 7;
      int p = tl * 16 + row;
      if (p < S2) out[((size_t)(bb * S2 + p)) * 256 + col] = v;
    }
  }
}

// ---------------------------------------------------------------------------
// Host launch
// ---------------------------------------------------------------------------
extern "C" void kernel_launch(void* const* d_in, const int* in_sizes, int n_in,
                              void* d_out, int out_size, void* d_ws, size_t ws_size,
                              hipStream_t stream) {
  (void)in_sizes; (void)n_in; (void)out_size; (void)ws_size;
  const float* input = (const float*)d_in[0];
  const float* masks = (const float*)d_in[1];
  const float* Wc    = (const float*)d_in[2];
  const float* bc    = (const float*)d_in[3];
  const float* Wr    = (const float*)d_in[4];
  const float* br    = (const float*)d_in[5];
  const float* Wof   = (const float*)d_in[6];
  const float* bof   = (const float*)d_in[7];
  const float* Wrf   = (const float*)d_in[8];
  const float* brf   = (const float*)d_in[9];
  const float* Wfc   = (const float*)d_in[10];
  const float* bfc   = (const float*)d_in[11];
  float* out = (float*)d_out;

  // --- workspace carve (all offsets 256B aligned) ---
  char* w = (char*)d_ws;
  auto carve = [&](size_t bytes) {
    void* p = (void*)w;
    w += (bytes + 255) & ~(size_t)255;
    return p;
  };
  bf16_t* X0h  = (bf16_t*)carve((size_t)NPIX * 256 * 2);
  bf16_t* ctxT = (bf16_t*)carve((size_t)NPIX * 256 * 2);
  bf16_t* relT = (bf16_t*)carve((size_t)NPIX * 384 * 2);
  bf16_t* P1h  = (bf16_t*)carve((size_t)NPIX * 256 * 2);
  bf16_t* P2h  = (bf16_t*)carve((size_t)NPIX * 256 * 2);
  float* P1f = (float*)carve((size_t)NPIX * 256 * 4);
  float* P2f = (float*)carve((size_t)NPIX * 256 * 4);
  float* Af  = (float*)carve((size_t)NPIX * 256 * 4);
  float* Bf  = (float*)carve((size_t)NPIX * 256 * 4);
  float* Cfe = (float*)carve((size_t)NPIX * 256 * 4);
  float* Df  = (float*)carve((size_t)NPIX * 256 * 4);
  float* Ef  = (float*)carve((size_t)NPIX * 256 * 4);
  float* cRel = (float*)carve(256 * 4);
  unsigned* Wb = (unsigned*)carve((size_t)KCHUNKS * 16 * 32 * 8 * 4); // 50.3 MB
  float* Prel = (float*)carve((size_t)32 * KSPLIT * 4096 * 4);
  float* Pobj = (float*)carve((size_t)4  * KSPLIT * 4096 * 4);

  // --- stage 0: conversions / constants ---
  k_cvt_in<<<NPIX, 256, 0, stream>>>(input, X0h);
  k_cvt_wfc<<<(KCHUNKS * 16 * 32 * 8) / 256, 256, 0, stream>>>(Wfc, Wb);
  k_crel<<<1, 256, 0, stream>>>(Wrf, brf, bof, cRel);

  // --- stage 1: per-pixel feature GEMM chain (all M = 1536) ---
  const int MT = NPIX / 16; // 96
  // ctx = X0 @ Wc^T + bc        -> ctxT (bf16)
  k_gemm<<<dim3(MT, 16), 32, 0, stream>>>(X0h, 256, Wc, 256, bc,
                                          nullptr, 0, ctxT, 256, 256);
  // rel = X0 @ Wr^T + br        -> relT (bf16, 384 ch)
  k_gemm<<<dim3(MT, 24), 32, 0, stream>>>(X0h, 256, Wr, 256, br,
                                          nullptr, 0, relT, 384, 256);
  // P1 = X0 @ WofA^T + ctx_lo @ WofB^T   -> P1f (f32) / P1h (bf16)
  k_gemm<<<dim3(MT, 16), 32, 0, stream>>>(X0h, 256, Wof, 512, nullptr,
                                          P1f, 0, nullptr, 0, 256);
  k_gemm<<<dim3(MT, 16), 32, 0, stream>>>(ctxT, 256, Wof + 256, 512, nullptr,
                                          P1f, 1, P1h, 256, 128);
  // P2 = ctx_hi @ WofC^T
  k_gemm<<<dim3(MT, 16), 32, 0, stream>>>(ctxT + 128, 256, Wof + 384, 512, nullptr,
                                          P2f, 0, P2h, 256, 128);
  // Af = P1 @ Wrf1^T + rel_x @ Wrf3^T
  k_gemm<<<dim3(MT, 16), 32, 0, stream>>>(P1h, 256, Wrf, 896, nullptr,
                                          Af, 0, nullptr, 0, 256);
  k_gemm<<<dim3(MT, 16), 32, 0, stream>>>(relT, 384, Wrf + 512, 896, nullptr,
                                          Af, 1, nullptr, 0, 128);
  // Bf = P1 @ Wrf2^T + rel_x @ Wrf3^T
  k_gemm<<<dim3(MT, 16), 32, 0, stream>>>(P1h, 256, Wrf + 256, 896, nullptr,
                                          Bf, 0, nullptr, 0, 256);
  k_gemm<<<dim3(MT, 16), 32, 0, stream>>>(relT, 384, Wrf + 512, 896, nullptr,
                                          Bf, 1, nullptr, 0, 128);
  // Cf = P2 @ Wrf1^T + rel_y @ Wrf4^T
  k_gemm<<<dim3(MT, 16), 32, 0, stream>>>(P2h, 256, Wrf, 896, nullptr,
                                          Cfe, 0, nullptr, 0, 256);
  k_gemm<<<dim3(MT, 16), 32, 0, stream>>>(relT + 128, 384, Wrf + 640, 896, nullptr,
                                          Cfe, 1, nullptr, 0, 128);
  // Df = P2 @ Wrf2^T + rel_z @ Wrf5^T
  k_gemm<<<dim3(MT, 16), 32, 0, stream>>>(P2h, 256, Wrf + 256, 896, nullptr,
                                          Df, 0, nullptr, 0, 256);
  k_gemm<<<dim3(MT, 16), 32, 0, stream>>>(relT + 256, 384, Wrf + 768, 896, nullptr,
                                          Df, 1, nullptr, 0, 128);
  // Ef = rel_y @ Wrf4^T + rel_z @ Wrf5^T
  k_gemm<<<dim3(MT, 16), 32, 0, stream>>>(relT + 128, 384, Wrf + 640, 896, nullptr,
                                          Ef, 0, nullptr, 0, 128);
  k_gemm<<<dim3(MT, 16), 32, 0, stream>>>(relT + 256, 384, Wrf + 768, 896, nullptr,
                                          Ef, 1, nullptr, 0, 128);

  // --- stage 2: fused relu + big GEMM vs Wfc (K split 8 ways) ---
  k_main<true ><<<dim3(4, KSPLIT), 256, 0, stream>>>(masks, P1f, P2f,
                                                     nullptr, nullptr, nullptr,
                                                     bof, Wb, Pobj);
  k_main<false><<<dim3(32, KSPLIT), 256, 0, stream>>>(masks, Af, Bf, Cfe, Df, Ef,
                                                      cRel, Wb, Prel);

  // --- stage 3: reduce partials, bias, normalize, write outputs ---
  k_fin<true ><<<4, 256, 0, stream>>>(Pobj, bfc, out);
  k_fin<false><<<32, 256, 0, stream>>>(Prel, bfc, out + OBJ_OUT_ELEMS);
}
